// MyMultiheadAttention_34359738574
// MI455X (gfx1250) — compile-verified, matrix operand order checked by reading, not run on hardware
//
#include <hip/hip_runtime.h>
#include <hip/hip_bf16.h>
#include <stdint.h>

// Problem constants (from reference)
#define D_MODEL   1024
#define NUM_HEAD  16
#define HEAD_DIM  64
#define SEQ       2048
#define BATCH     2
#define MROWS     (SEQ * BATCH)   // 4096 flattened (s,b) rows
#define QSET      2               // 2 x 16 = 32 query rows per wave
#define KTILE     64              // keys per online-softmax update

typedef __bf16 bf16_t;
typedef bf16_t v16bf __attribute__((ext_vector_type(16)));
typedef float  v8f   __attribute__((ext_vector_type(8)));

struct alignas(16) U128 { unsigned int x[4]; };
struct U256 { U128 lo, hi; };

// fp32 -> bf16 round-to-nearest-even
__device__ __forceinline__ unsigned short f2bf(float f) {
  unsigned int u = __builtin_bit_cast(unsigned int, f);
  unsigned int r = u + 0x7FFFu + ((u >> 16) & 1u);
  return (unsigned short)(r >> 16);
}

// Load one 16x32 bf16 WMMA A/B fragment. Per the CDNA5 ISA 16-bit fragment
// layout, lane (half = lane>>4, idx = lane&15) holds two contiguous 16-byte
// runs of K: [half*8 .. half*8+7] and [16+half*8 .. 16+half*8+7], for matrix
// row/col (row + idx). K must be the contiguous axis of `base` with leading
// dimension `ld` (elements).
__device__ __forceinline__ v16bf load_frag(const unsigned short* base, int row,
                                           int ld, int k0, int lane) {
  const int half = (lane >> 4) & 1;
  const int idx  = lane & 15;
  const unsigned short* p = base + (size_t)(row + idx) * ld + k0 + half * 8;
  U256 t;
  t.lo = *(const U128*)(p);
  t.hi = *(const U128*)(p + 16);
  return __builtin_bit_cast(v16bf, t);
}

__device__ __forceinline__ v8f wmma_bf16f32(v16bf a, v16bf b, v8f c) {
  // (neg_a, A, neg_b, B, c_mod, C, reuse_a, reuse_b)
  return __builtin_amdgcn_wmma_f32_16x16x32_bf16(false, a, false, b, (short)0, c,
                                                 false, false);
}

__device__ __forceinline__ void sched_fence() {
#if __has_builtin(__builtin_amdgcn_wave_barrier)
  __builtin_amdgcn_wave_barrier();
#else
  __asm__ volatile("" ::: "memory");
#endif
}

// ---- wave32 16-lane row reductions via DPP ROW_ROR (pure VALU, no LDS) ----
// DPP16 rows are lanes 0-15 / 16-31, exactly matching the C-fragment halves.
// Rotates by 1,2,4,8 form an all-reduce over each 16-lane row.
template <int CTRL>
__device__ __forceinline__ float dpp_rot(float x) {
  return __builtin_bit_cast(
      float, __builtin_amdgcn_update_dpp(0, __builtin_bit_cast(int, x), CTRL,
                                         0xF, 0xF, true));
}
__device__ __forceinline__ float rowmax16(float x) {
  x = fmaxf(x, dpp_rot<0x121>(x));  // ROW_ROR:1
  x = fmaxf(x, dpp_rot<0x122>(x));  // ROW_ROR:2
  x = fmaxf(x, dpp_rot<0x124>(x));  // ROW_ROR:4
  x = fmaxf(x, dpp_rot<0x128>(x));  // ROW_ROR:8
  return x;
}
__device__ __forceinline__ float rowsum16(float x) {
  x += dpp_rot<0x121>(x);
  x += dpp_rot<0x122>(x);
  x += dpp_rot<0x124>(x);
  x += dpp_rot<0x128>(x);
  return x;
}

// ---------------------------------------------------------------- cast kernel
__global__ void mha_cast_bf16(const float4* __restrict__ in,
                              ushort4* __restrict__ out, int n4) {
  int i = blockIdx.x * blockDim.x + threadIdx.x;
  int stride = gridDim.x * blockDim.x;
  for (; i < n4; i += stride) {
    float4 f = in[i];
    ushort4 u;
    u.x = f2bf(f.x); u.y = f2bf(f.y); u.z = f2bf(f.z); u.w = f2bf(f.w);
    out[i] = u;
  }
}

// ------------------------------------------------------------------ GEMM
// out(m,n) = sum_k A[m,k] * W[n,k] + bias[n]   (torch Linear: x @ W.T + b)
// MODE 0: bf16 out, row-major (M,N)
// MODE 1: bf16 out, V-transposed layout (h,b,d,s) for the attention B-operand
// MODE 2: fp32 out, row-major (M,N)  (final projection)
template <int MODE>
__global__ __launch_bounds__(128)
void mha_gemm_wmma(const unsigned short* __restrict__ A,
                   const unsigned short* __restrict__ W,
                   const float* __restrict__ bias,
                   void* __restrict__ outp, int M, int N, int K) {
  const int lane = threadIdx.x & 31;
  const int wave = threadIdx.x >> 5;
  const int row0 = blockIdx.y * 64 + wave * 16;  // 4 waves stacked along M
  const int col0 = blockIdx.x * 64;              // each wave: 16M x 64N tile

  v8f acc0 = {}, acc1 = {}, acc2 = {}, acc3 = {};

  for (int k0 = 0; k0 < K; k0 += 32) {
    if (k0 + 32 < K) {  // hint the next A stripe into cache
      __builtin_prefetch(A + (size_t)(row0 + (lane & 15)) * K + k0 + 32, 0, 1);
    }
    v16bf a  = load_frag(A, row0, K, k0, lane);
    v16bf b0 = load_frag(W, col0 +  0, K, k0, lane);
    v16bf b1 = load_frag(W, col0 + 16, K, k0, lane);
    v16bf b2 = load_frag(W, col0 + 32, K, k0, lane);
    v16bf b3 = load_frag(W, col0 + 48, K, k0, lane);
    acc0 = wmma_bf16f32(a, b0, acc0);
    acc1 = wmma_bf16f32(a, b1, acc1);
    acc2 = wmma_bf16f32(a, b2, acc2);
    acc3 = wmma_bf16f32(a, b3, acc3);
  }

  const int ncol  = lane & 15;
  const int mbase = (lane >> 4) * 8;  // C/D layout: VGPR r -> M = r + 8*half
  v8f accs[4] = {acc0, acc1, acc2, acc3};
#pragma unroll
  for (int nt = 0; nt < 4; ++nt) {
    const int n  = col0 + nt * 16 + ncol;
    const float bv = bias[n];
#pragma unroll
    for (int r = 0; r < 8; ++r) {
      const int m = row0 + mbase + r;
      const float val = accs[nt][r] + bv;
      if (MODE == 0) {
        ((unsigned short*)outp)[(size_t)m * N + n] = f2bf(val);
      } else if (MODE == 1) {
        const int s = m >> 1, bb = m & 1;          // m = s*BATCH + b
        const int h = n >> 6, d  = n & 63;         // n = h*HEAD_DIM + d
        ((unsigned short*)outp)[(((size_t)h * BATCH + bb) * HEAD_DIM + d) * SEQ + s] =
            f2bf(val);
      } else {
        ((float*)outp)[(size_t)m * N + n] = val;
      }
    }
  }
}

// ------------------------------------------------------------- attention
// One block = one (h,b) pair x 128 query rows; 4 waves x 32 rows (2 row-sets).
// Both row-sets share K/V fragments (halves K/V L2 traffic); 64-key tiles so
// each online-softmax update (DPP reductions + O rescale) amortizes over
// twice as many WMMAs. Flash-style, never materializes the 2048^2 scores.
__global__ __launch_bounds__(128)
void mha_attn_wmma(const unsigned short* __restrict__ Qp,  // (s,b,D) bf16
                   const unsigned short* __restrict__ Kp,  // (s,b,D) bf16
                   const unsigned short* __restrict__ Vt,  // (h,b,d,s) bf16
                   const float* __restrict__ mask,         // (S,S) fp32
                   unsigned short* __restrict__ Obf) {     // (h,b,s,dh) bf16
  __shared__ unsigned short Plds[4][QSET][16 * KTILE];  // per-wave P staging

  const int lane = threadIdx.x & 31;
  const int wave = threadIdx.x >> 5;
  const int q0 = blockIdx.x * (16 * QSET * 4) + wave * (16 * QSET);
  const int hb = blockIdx.y;
  const int h = hb >> 1, b = hb & 1;

  const unsigned short* Qb = Qp + (size_t)b * D_MODEL + (size_t)h * HEAD_DIM;
  const unsigned short* Kb = Kp + (size_t)b * D_MODEL + (size_t)h * HEAD_DIM;
  const unsigned short* Vb = Vt + ((size_t)h * BATCH + b) * HEAD_DIM * SEQ;
  const int ldq = BATCH * D_MODEL;  // stride between successive seq rows

  const v8f vzero = {};

  // Q fragments: QSET row-sets x (Dh=64 -> two K=32 A-fragments), resident
  v16bf aq[QSET][2];
#pragma unroll
  for (int rs = 0; rs < QSET; ++rs) {
    aq[rs][0] = load_frag(Qb, q0 + rs * 16, ldq, 0,  lane);
    aq[rs][1] = load_frag(Qb, q0 + rs * 16, ldq, 32, lane);
  }

  v8f o[QSET][4];
  float rmax[QSET][8], rsum[QSET][8];
#pragma unroll
  for (int rs = 0; rs < QSET; ++rs) {
#pragma unroll
    for (int nt = 0; nt < 4; ++nt) o[rs][nt] = vzero;
#pragma unroll
    for (int r = 0; r < 8; ++r) { rmax[rs][r] = -3.0e38f; rsum[rs][r] = 0.0f; }
  }

  const int ncol = lane & 15;
  const int mb   = (lane >> 4) * 8;
  const float scale = 0.125f;  // 1/sqrt(64)

  for (int k0 = 0; k0 < SEQ; k0 += KTILE) {
    // K fragments for this 64-key tile (4 j-subtiles x 2 d-halves), shared
    v16bf bk[8];
#pragma unroll
    for (int jt = 0; jt < 4; ++jt) {
      bk[jt * 2 + 0] = load_frag(Kb, k0 + jt * 16, ldq, 0,  lane);
      bk[jt * 2 + 1] = load_frag(Kb, k0 + jt * 16, ldq, 32, lane);
    }

#pragma unroll
    for (int rs = 0; rs < QSET; ++rs) {
      // scores S[16 x 64] = Q(16x64) . K(64x64)^T
      v8f s[4];
#pragma unroll
      for (int jt = 0; jt < 4; ++jt) {
        s[jt] = wmma_bf16f32(aq[rs][0], bk[jt * 2 + 0], vzero);
        s[jt] = wmma_bf16f32(aq[rs][1], bk[jt * 2 + 1], s[jt]);
      }

      // one online-softmax update per 64 keys: C-fragment rows
      // (VGPR r <-> M = r + 8*half) align with per-lane row stats.
#pragma unroll
      for (int r = 0; r < 8; ++r) {
        const int m = q0 + rs * 16 + mb + r;
        const float* mrow = mask + (size_t)m * SEQ + k0 + ncol;
        float x[4];
#pragma unroll
        for (int jt = 0; jt < 4; ++jt) x[jt] = s[jt][r] * scale + mrow[jt * 16];
        float mx = fmaxf(fmaxf(x[0], x[1]), fmaxf(x[2], x[3]));
        mx = rowmax16(mx);
        const float nm   = fmaxf(rmax[rs][r], mx);
        const float corr = __expf(rmax[rs][r] - nm);
        rmax[rs][r] = nm;
        float p[4];
#pragma unroll
        for (int jt = 0; jt < 4; ++jt) p[jt] = __expf(x[jt] - nm);
        const float ls = rowsum16((p[0] + p[1]) + (p[2] + p[3]));
        rsum[rs][r] = rsum[rs][r] * corr + ls;
        o[rs][0][r] *= corr; o[rs][1][r] *= corr;
        o[rs][2][r] *= corr; o[rs][3][r] *= corr;
        // stage P (C layout) into LDS row-major for A-fragment re-read
#pragma unroll
        for (int jt = 0; jt < 4; ++jt)
          Plds[wave][rs][(mb + r) * KTILE + jt * 16 + ncol] = f2bf(p[jt]);
      }
    }
    sched_fence();  // same-wave DS ops are in-order; fence compiler scheduling
    v16bf ap[QSET][2];
#pragma unroll
    for (int rs = 0; rs < QSET; ++rs) {
      ap[rs][0] = load_frag(&Plds[wave][rs][0], 0, KTILE, 0,  lane);
      ap[rs][1] = load_frag(&Plds[wave][rs][0], 0, KTILE, 32, lane);
    }
    sched_fence();

    // O += P(16x64) . V(64x64); V fragments shared by both row-sets
#pragma unroll
    for (int nt = 0; nt < 4; ++nt) {
      const v16bf bv0 = load_frag(Vb, nt * 16, SEQ, k0,      lane);
      const v16bf bv1 = load_frag(Vb, nt * 16, SEQ, k0 + 32, lane);
#pragma unroll
      for (int rs = 0; rs < QSET; ++rs) {
        o[rs][nt] = wmma_bf16f32(ap[rs][0], bv0, o[rs][nt]);
        o[rs][nt] = wmma_bf16f32(ap[rs][1], bv1, o[rs][nt]);
      }
    }
  }

  // normalize and store in (h,b,s,dh)-contiguous order (reference's raw view)
  unsigned short* Ob = Obf + ((size_t)h * BATCH + b) * SEQ * HEAD_DIM;
#pragma unroll
  for (int rs = 0; rs < QSET; ++rs) {
#pragma unroll
    for (int nt = 0; nt < 4; ++nt) {
#pragma unroll
      for (int r = 0; r < 8; ++r) {
        const float inv = 1.0f / rsum[rs][r];
        Ob[(size_t)(q0 + rs * 16 + mb + r) * HEAD_DIM + nt * 16 + ncol] =
            f2bf(o[rs][nt][r] * inv);
      }
    }
  }
}

// ---------------------------------------------------------------- launcher
extern "C" void kernel_launch(void* const* d_in, const int* in_sizes, int n_in,
                              void* d_out, int out_size, void* d_ws, size_t ws_size,
                              hipStream_t stream) {
  (void)in_sizes; (void)n_in; (void)out_size; (void)ws_size;

  const float* q    = (const float*)d_in[0];
  const float* k    = (const float*)d_in[1];
  const float* v    = (const float*)d_in[2];
  const float* mask = (const float*)d_in[3];
  const float* Wq   = (const float*)d_in[4];
  const float* bq   = (const float*)d_in[5];
  const float* Wk   = (const float*)d_in[6];
  const float* bk   = (const float*)d_in[7];
  const float* Wv   = (const float*)d_in[8];
  const float* bv   = (const float*)d_in[9];
  const float* Wo   = (const float*)d_in[10];
  const float* bo   = (const float*)d_in[11];

  const size_t SZ_X = (size_t)MROWS * D_MODEL;    // 4M elements
  const size_t SZ_W = (size_t)D_MODEL * D_MODEL;  // 1M elements
  unsigned short* ws  = (unsigned short*)d_ws;    // all bf16 scratch, ~64 MB
  unsigned short* Xq  = ws;
  unsigned short* Xk  = Xq  + SZ_X;
  unsigned short* Xv  = Xk  + SZ_X;
  unsigned short* Wqb = Xv  + SZ_X;
  unsigned short* Wkb = Wqb + SZ_W;
  unsigned short* Wvb = Wkb + SZ_W;
  unsigned short* Wob = Wvb + SZ_W;
  unsigned short* Qp  = Wob + SZ_W;  // (s,b,D)
  unsigned short* Kp  = Qp  + SZ_X;  // (s,b,D)
  unsigned short* Vt  = Kp  + SZ_X;  // (h,b,d,s)
  unsigned short* Ao  = Vt  + SZ_X;  // (h,b,s,dh)

  auto cast = [&](const float* src, unsigned short* dst, size_t n) {
    const unsigned n4 = (unsigned)(n / 4);
    mha_cast_bf16<<<dim3((n4 + 255) / 256), 256, 0, stream>>>(
        (const float4*)src, (ushort4*)dst, (int)n4);
  };
  cast(q,  Xq,  SZ_X);
  cast(k,  Xk,  SZ_X);
  cast(v,  Xv,  SZ_X);
  cast(Wq, Wqb, SZ_W);
  cast(Wk, Wkb, SZ_W);
  cast(Wv, Wvb, SZ_W);
  cast(Wo, Wob, SZ_W);

  dim3 gb(D_MODEL / 64, MROWS / 64);  // (16, 64), 128 threads = 4 waves
  mha_gemm_wmma<0><<<gb, 128, 0, stream>>>(Xq, Wqb, bq, Qp, MROWS, D_MODEL, D_MODEL);
  mha_gemm_wmma<0><<<gb, 128, 0, stream>>>(Xk, Wkb, bk, Kp, MROWS, D_MODEL, D_MODEL);
  mha_gemm_wmma<1><<<gb, 128, 0, stream>>>(Xv, Wvb, bv, Vt, MROWS, D_MODEL, D_MODEL);

  dim3 ga(SEQ / 128, NUM_HEAD * BATCH);  // (16, 32)
  mha_attn_wmma<<<ga, 128, 0, stream>>>(Qp, Kp, Vt, mask, Ao);

  mha_gemm_wmma<2><<<gb, 128, 0, stream>>>(Ao, Wob, bo, d_out, MROWS, D_MODEL, D_MODEL);
}
